// GATBlock_3272765079675
// MI455X (gfx1250) — matrix operand order repaired and checked by reading, not compiled, not run on hardware
//
#include <hip/hip_runtime.h>
#include <hip/hip_bf16.h>
#include <stdint.h>

// GATv2 block for MI455X (gfx1250, wave32, WMMA + async LDS staging).
// N=50000 nodes, E=800000 edges, D=128, H=8 heads, C=16 channels/head.
#define NN 50000
#define EE 800000
#define DD 128
#define HH 8
#define CC 16
#define DFF 512
#define NT_EDGE 8  // 16-edge tiles per block in the fused edge kernel

static constexpr float NEG_SLOPE = 0.2f;
static constexpr float LN_EPS = 1e-5f;

typedef __bf16 bf16;
typedef __bf16 v16bf __attribute__((ext_vector_type(16)));
typedef __bf16 v8bf  __attribute__((ext_vector_type(8)));
typedef float  v8f   __attribute__((ext_vector_type(8)));
typedef int    v4i_t __attribute__((vector_size(16)));  // matches async-LDS builtin param

// ------------------------------------------------ async global->LDS staging

#if __has_builtin(__builtin_amdgcn_global_load_async_to_lds_b128)
#define HAVE_ASYNC 1
#else
#define HAVE_ASYNC 0
#endif
#if __has_builtin(__builtin_amdgcn_s_wait_asynccnt)
#define HAVE_WAIT_ASYNC 1
#else
#define HAVE_WAIT_ASYNC 0
#endif

// Copy 16 bytes global -> LDS. Async (ASYNCcnt-tracked DMA) when available.
__device__ __forceinline__ void stage16(void* lds, const void* g) {
#if HAVE_ASYNC
  __builtin_amdgcn_global_load_async_to_lds_b128(
      (__attribute__((address_space(1))) v4i_t*)g,
      (__attribute__((address_space(3))) v4i_t*)lds, 0, 0);
#else
  *(uint4*)lds = *(const uint4*)g;
#endif
}

template <int N>
__device__ __forceinline__ void stage_wait() {
#if HAVE_ASYNC
#if HAVE_WAIT_ASYNC
  __builtin_amdgcn_s_wait_asynccnt(N);
#else
  asm volatile("s_wait_asynccnt %0" ::"i"(N) : "memory");
#endif
#endif
}

// ---------------------------------------------------------------- helpers

__device__ __forceinline__ v8f wmma_bf16(v16bf a, v16bf b, v8f c) {
  return __builtin_amdgcn_wmma_f32_16x16x32_bf16(
      /*neg_a=*/false, a, /*neg_b=*/false, b,
      /*c_mod=*/(short)0, c, /*reuse_a=*/false, /*reuse_b=*/false);
}

// A-fragment (16x32 bf16) from an LDS bf16 tile [16 x rowlen].
__device__ __forceinline__ v16bf load_a_lds(const bf16* s, int rowlen, int kw, int lane) {
  int kb = kw + ((lane < 16) ? 0 : 8);
  const bf16* p = s + (lane & 15) * rowlen + kb;
  v8bf lo = *(const v8bf*)p;
  v8bf hi = *(const v8bf*)(p + 16);
  v16bf a;
#pragma unroll
  for (int i = 0; i < 8; ++i) { a[i] = lo[i]; a[8 + i] = hi[i]; }
  return a;
}

// A-fragment from an LDS fp32 tile, converting to bf16 in-lane
// (v_cvt co-executes with the matrix pipe).
__device__ __forceinline__ v16bf load_a_lds_f32(const float* s, int rowlen, int kw, int lane) {
  int kb = kw + ((lane < 16) ? 0 : 8);
  const float* p = s + (lane & 15) * rowlen + kb;
  float4 a0 = *(const float4*)p;
  float4 a1 = *(const float4*)(p + 4);
  float4 a2 = *(const float4*)(p + 16);
  float4 a3 = *(const float4*)(p + 20);
  v16bf a;
  a[0] = (bf16)a0.x;  a[1] = (bf16)a0.y;  a[2] = (bf16)a0.z;  a[3] = (bf16)a0.w;
  a[4] = (bf16)a1.x;  a[5] = (bf16)a1.y;  a[6] = (bf16)a1.z;  a[7] = (bf16)a1.w;
  a[8] = (bf16)a2.x;  a[9] = (bf16)a2.y;  a[10] = (bf16)a2.z; a[11] = (bf16)a2.w;
  a[12] = (bf16)a3.x; a[13] = (bf16)a3.y; a[14] = (bf16)a3.z; a[15] = (bf16)a3.w;
  return a;
}

// B-fragment (32x16 bf16) for y = x @ W^T: k = kw+8h..+7 and kw+16+8h..+7 of weight row n.
__device__ __forceinline__ v16bf load_b_w(const bf16* w, int nrow, int rowlen, int kw, int lane) {
  const bf16* p = w + (size_t)nrow * rowlen + kw + ((lane < 16) ? 0 : 8);
  v8bf lo = *(const v8bf*)p;  // 16B aligned
  v8bf hi = *(const v8bf*)(p + 16);
  v16bf b;
#pragma unroll
  for (int i = 0; i < 8; ++i) { b[i] = lo[i]; b[8 + i] = hi[i]; }
  return b;
}

__device__ __forceinline__ float wave_allred(float v) {
  v += __shfl_xor(v, 16);
  v += __shfl_xor(v, 8);
  v += __shfl_xor(v, 4);
  v += __shfl_xor(v, 2);
  v += __shfl_xor(v, 1);
  return v;
}

__device__ __forceinline__ unsigned f2ord(float f) {
  unsigned u = __float_as_uint(f);
  return (u & 0x80000000u) ? ~u : (u | 0x80000000u);
}
__device__ __forceinline__ float ord2f(unsigned u) {
  unsigned v = (u & 0x80000000u) ? (u & 0x7FFFFFFFu) : ~u;
  return __uint_as_float(v);
}

// ---------------------------------------------------------------- kernels

__global__ void k_cvt_bf16(const float* __restrict__ src, bf16* __restrict__ dst, int n) {
  int i = (blockIdx.x * blockDim.x + threadIdx.x) * 4;
  if (i + 3 < n) {
    float4 f = *(const float4*)(src + i);
    alignas(8) bf16 r[4] = {(bf16)f.x, (bf16)f.y, (bf16)f.z, (bf16)f.w};
    *(uint2*)(dst + i) = *(const uint2*)r;
  } else {
    for (int j = i; j < n; ++j) dst[j] = (bf16)src[j];
  }
}

// h = LayerNorm(x) * ln1_w -> bf16. One wave per row.
__global__ void k_ln1(const float* __restrict__ x, const float* __restrict__ g,
                      bf16* __restrict__ out) {
  int row = (blockIdx.x * blockDim.x + threadIdx.x) >> 5;
  int lane = threadIdx.x & 31;
  if (row >= NN) return;
  int c0 = lane * 4;
  float4 v = *(const float4*)(x + (size_t)row * DD + c0);
  float mu = wave_allred(v.x + v.y + v.z + v.w) * (1.0f / DD);
  float dx = v.x - mu, dy = v.y - mu, dz = v.z - mu, dw = v.w - mu;
  float var = wave_allred(dx * dx + dy * dy + dz * dz + dw * dw) * (1.0f / DD);
  float inv = rsqrtf(var + LN_EPS);
  alignas(8) bf16 r[4] = {(bf16)(dx * inv * g[c0 + 0]), (bf16)(dy * inv * g[c0 + 1]),
                          (bf16)(dz * inv * g[c0 + 2]), (bf16)(dw * inv * g[c0 + 3])};
  *(uint2*)(out + (size_t)row * DD + c0) = *(const uint2*)r;
}

// xl / xr = h @ W^T + b (blockIdx.y selects l/r).
__global__ void k_xlxr(const bf16* __restrict__ h,
                       const bf16* __restrict__ wl, const float* __restrict__ bl,
                       float* __restrict__ xl,
                       const bf16* __restrict__ wr, const float* __restrict__ br,
                       float* __restrict__ xr) {
  __shared__ bf16 S[16 * DD];
  const bf16* w = blockIdx.y ? wr : wl;
  const float* bias = blockIdx.y ? br : bl;
  float* out = blockIdx.y ? xr : xl;
  int row0 = blockIdx.x * 16;
  int tid = threadIdx.x;
  stage16(&S[tid * 8], h + (size_t)row0 * DD + tid * 8);
  stage_wait<0>();
  __syncthreads();
  int wv = tid >> 5, lane = tid & 31;
  int nrow = wv * 16 + (lane & 15);
  v8f acc = {0, 0, 0, 0, 0, 0, 0, 0};
#pragma unroll
  for (int kk = 0; kk < 4; ++kk)
    acc = wmma_bf16(load_a_lds(S, DD, kk * 32, lane), load_b_w(w, nrow, DD, kk * 32, lane), acc);
  float bv = bias[nrow];
  int mbase = (lane < 16) ? 0 : 8;
#pragma unroll
  for (int r = 0; r < 8; ++r)
    out[(size_t)(row0 + r + mbase) * DD + nrow] = acc[r] + bv;
}

// Fused edge pass. Each block: NT_EDGE tiles of 16 edges, double-buffered
// async staging of raw fp32 edge_attr rows into LDS; wave == head (C==16);
//   z = leaky_relu((edge_attr@We^T)[:,head] + xl[src] + xr[dst])
//   logit = z . att[head]  (cross-lane reduce) ; segment-max via atomicMax.
// B-fragments of We and att are hoisted: loaded once per block.
__global__ void k_edge(const float* __restrict__ eattr, const int* __restrict__ src,
                       const int* __restrict__ dst, const bf16* __restrict__ we,
                       const float* __restrict__ xl, const float* __restrict__ xr,
                       const float* __restrict__ att, float* __restrict__ logits,
                       unsigned* __restrict__ mbits) {
  __shared__ float Af[2][16 * DD];   // raw fp32 edge rows, 2 x 8KB
  __shared__ int sIdx[2][32];        // [0..15]=src, [16..31]=dst
  int tid = threadIdx.x;
  int wv = tid >> 5, lane = tid & 31;  // wv == head
  int tile0 = blockIdx.x * NT_EDGE;

  int n = wv * CC + (lane & 15);  // absolute output channel
  v16bf b0 = load_b_w(we, n, DD, 0, lane);
  v16bf b1 = load_b_w(we, n, DD, 32, lane);
  v16bf b2 = load_b_w(we, n, DD, 64, lane);
  v16bf b3 = load_b_w(we, n, DD, 96, lane);
  float aw = att[wv * CC + (lane & 15)];
  int mbase = (lane < 16) ? 0 : 8;

  auto issue = [&](int t, int buf) {
    int e0 = (tile0 + t) * 16;
    const float* base = eattr + (size_t)e0 * DD + tid * 8;
    stage16(&Af[buf][tid * 8], base);
    stage16(&Af[buf][tid * 8 + 4], base + 4);
    if (tid < 32) sIdx[buf][tid] = (tid < 16) ? src[e0 + tid] : dst[e0 + tid - 16];
  };

  issue(0, 0);
  for (int t = 0; t < NT_EDGE; ++t) {
    int buf = t & 1;
    if (t + 1 < NT_EDGE) {
      issue(t + 1, buf ^ 1);
      stage_wait<2>();  // in-order: the 2 older loads (this tile) are done
    } else {
      stage_wait<0>();
    }
    __syncthreads();

    const float* A = Af[buf];
    v8f acc = {0, 0, 0, 0, 0, 0, 0, 0};
    acc = wmma_bf16(load_a_lds_f32(A, DD, 0, lane), b0, acc);
    acc = wmma_bf16(load_a_lds_f32(A, DD, 32, lane), b1, acc);
    acc = wmma_bf16(load_a_lds_f32(A, DD, 64, lane), b2, acc);
    acc = wmma_bf16(load_a_lds_f32(A, DD, 96, lane), b3, acc);

    int e0 = (tile0 + t) * 16;
#pragma unroll
    for (int r = 0; r < 8; ++r) {
      int M = r + mbase;
      int sN = sIdx[buf][M], dN = sIdx[buf][16 + M];
      float z = acc[r] + xl[(size_t)sN * DD + n] + xr[(size_t)dN * DD + n];
      z = (z > 0.0f) ? z : z * NEG_SLOPE;
      float v = z * aw;
      v += __shfl_xor(v, 1);
      v += __shfl_xor(v, 2);
      v += __shfl_xor(v, 4);
      v += __shfl_xor(v, 8);  // reduce within each 16-lane half
      if ((lane & 15) == 0) {
        logits[(size_t)(e0 + M) * HH + wv] = v;
        atomicMax(&mbits[(size_t)dN * HH + wv], f2ord(v));
      }
    }
    __syncthreads();  // everyone done reading buf before it is refilled
  }
}

// num = exp(logit - m[dst]); den[dst] += num.  (E*H threads)
__global__ void k_expden(const int* __restrict__ dst, float* __restrict__ logits,
                         const unsigned* __restrict__ mbits, float* __restrict__ den) {
  int i = blockIdx.x * blockDim.x + threadIdx.x;
  int e = i >> 3, h = i & 7;
  int d = dst[e];
  float m = ord2f(mbits[(size_t)d * HH + h]);
  float num = __expf(logits[(size_t)e * HH + h] - m);
  logits[(size_t)e * HH + h] = num;
  atomicAdd(&den[(size_t)d * HH + h], num);
}

// attn_out[dst] += xl[src] * alpha.  (E*32 threads, 4 channels each)
__global__ void k_msg(const int* __restrict__ src, const int* __restrict__ dst,
                      const float* __restrict__ num, const float* __restrict__ den,
                      const float* __restrict__ xl, float* __restrict__ attn) {
  int i = blockIdx.x * blockDim.x + threadIdx.x;
  int e = i >> 5, q = i & 31;
  int c0 = q * 4, h = q >> 2;
  int s = src[e], d = dst[e];
  float alpha = num[(size_t)e * HH + h] / den[(size_t)d * HH + h];
  float4 v = *(const float4*)(xl + (size_t)s * DD + c0);
  float* o = attn + (size_t)d * DD + c0;
  atomicAdd(o + 0, v.x * alpha);
  atomicAdd(o + 1, v.y * alpha);
  atomicAdd(o + 2, v.z * alpha);
  atomicAdd(o + 3, v.w * alpha);
}

// x2 = x + attn_out + attn_bias (fp32); h2 = LN(x2)*ln2_w -> bf16.
__global__ void k_resid_ln2(const float* __restrict__ x, const float* __restrict__ attn,
                            const float* __restrict__ ab, const float* __restrict__ g,
                            float* __restrict__ x2, bf16* __restrict__ h2) {
  int row = (blockIdx.x * blockDim.x + threadIdx.x) >> 5;
  int lane = threadIdx.x & 31;
  if (row >= NN) return;
  int c0 = lane * 4;
  float4 v = *(const float4*)(x + (size_t)row * DD + c0);
  float4 a = *(const float4*)(attn + (size_t)row * DD + c0);
  float4 b = *(const float4*)(ab + c0);
  v.x += a.x + b.x; v.y += a.y + b.y; v.z += a.z + b.z; v.w += a.w + b.w;
  *(float4*)(x2 + (size_t)row * DD + c0) = v;
  float mu = wave_allred(v.x + v.y + v.z + v.w) * (1.0f / DD);
  float dx = v.x - mu, dy = v.y - mu, dz = v.z - mu, dw = v.w - mu;
  float var = wave_allred(dx * dx + dy * dy + dz * dz + dw * dw) * (1.0f / DD);
  float inv = rsqrtf(var + LN_EPS);
  alignas(8) bf16 r[4] = {(bf16)(dx * inv * g[c0 + 0]), (bf16)(dy * inv * g[c0 + 1]),
                          (bf16)(dz * inv * g[c0 + 2]), (bf16)(dw * inv * g[c0 + 3])};
  *(uint2*)(h2 + (size_t)row * DD + c0) = *(const uint2*)r;
}

// t = relu(h2 @ fc_w^T) -> bf16 [N,512]. 8 waves x 4 col tiles, K=128.
__global__ void k_mlp1(const bf16* __restrict__ h2, const bf16* __restrict__ wfc,
                       bf16* __restrict__ t) {
  __shared__ bf16 S[16 * DD];
  int row0 = blockIdx.x * 16;
  int tid = threadIdx.x;
  stage16(&S[tid * 8], h2 + (size_t)row0 * DD + tid * 8);
  stage_wait<0>();
  __syncthreads();
  int wv = tid >> 5, lane = tid & 31;
  int mbase = (lane < 16) ? 0 : 8;
#pragma unroll
  for (int tt = 0; tt < 4; ++tt) {
    int nrow = wv * 64 + tt * 16 + (lane & 15);
    v8f acc = {0, 0, 0, 0, 0, 0, 0, 0};
#pragma unroll
    for (int kk = 0; kk < 4; ++kk)
      acc = wmma_bf16(load_a_lds(S, DD, kk * 32, lane), load_b_w(wfc, nrow, DD, kk * 32, lane), acc);
#pragma unroll
    for (int r = 0; r < 8; ++r) {
      float v = acc[r];
      t[(size_t)(row0 + r + mbase) * DFF + nrow] = (bf16)(v > 0.0f ? v : 0.0f);
    }
  }
}

// out = x2 + t @ proj_w^T. K=512 -> 16 WMMAs per wave; 16KB LDS tile of t.
__global__ void k_mlp2(const bf16* __restrict__ t, const bf16* __restrict__ wpr,
                       const float* __restrict__ x2, float* __restrict__ out) {
  __shared__ bf16 S[16 * DFF];
  int row0 = blockIdx.x * 16;
  int tid = threadIdx.x;
#pragma unroll
  for (int j = 0; j < 4; ++j)
    stage16(&S[tid * 32 + j * 8], t + (size_t)row0 * DFF + tid * 32 + j * 8);
  stage_wait<0>();
  __syncthreads();
  int wv = tid >> 5, lane = tid & 31;
  int nrow = wv * 16 + (lane & 15);
  v8f acc = {0, 0, 0, 0, 0, 0, 0, 0};
#pragma unroll
  for (int kk = 0; kk < 16; ++kk)
    acc = wmma_bf16(load_a_lds(S, DFF, kk * 32, lane), load_b_w(wpr, nrow, DFF, kk * 32, lane), acc);
  int mbase = (lane < 16) ? 0 : 8;
#pragma unroll
  for (int r = 0; r < 8; ++r) {
    int row = row0 + r + mbase;
    out[(size_t)row * DD + nrow] = x2[(size_t)row * DD + nrow] + acc[r];
  }
}

// ---------------------------------------------------------------- launch

extern "C" void kernel_launch(void* const* d_in, const int* in_sizes, int n_in,
                              void* d_out, int out_size, void* d_ws, size_t ws_size,
                              hipStream_t stream) {
  (void)in_sizes; (void)n_in; (void)out_size; (void)ws_size;
  const float* x       = (const float*)d_in[0];
  const int*   ei      = (const int*)d_in[1];  // [2,E]: src=ei, dst=ei+E
  const float* eattr   = (const float*)d_in[2];
  const float* ln1_w   = (const float*)d_in[3];
  const float* lin_l_w = (const float*)d_in[4];
  const float* lin_l_b = (const float*)d_in[5];
  const float* lin_r_w = (const float*)d_in[6];
  const float* lin_r_b = (const float*)d_in[7];
  const float* lin_e_w = (const float*)d_in[8];
  const float* att     = (const float*)d_in[9];
  const float* attn_b  = (const float*)d_in[10];
  const float* ln2_w   = (const float*)d_in[11];
  const float* fc_w    = (const float*)d_in[12];
  const float* proj_w  = (const float*)d_in[13];
  float* out = (float*)d_out;

  char* ws = (char*)d_ws;
  size_t off = 0;
  auto alloc = [&](size_t bytes) -> char* {
    char* p = ws + off;
    off = (off + bytes + 255) & ~(size_t)255;
    return p;
  };
  bf16*     h_bf   = (bf16*)alloc((size_t)NN * DD * 2);   // reused as h2
  float*    xl     = (float*)alloc((size_t)NN * DD * 4);
  float*    xr     = (float*)alloc((size_t)NN * DD * 4);  // reused as x2
  char*     Rbase  = ws + off;                            // union region (-> t)
  float*    logits = (float*)alloc((size_t)EE * HH * 4);  // then holds num
  unsigned* mb     = (unsigned*)alloc((size_t)NN * HH * 4);
  float*    den    = (float*)alloc((size_t)NN * HH * 4);
  float*    attn   = (float*)alloc((size_t)NN * DD * 4);
  bf16*     wlb    = (bf16*)alloc((size_t)DD * DD * 2);
  bf16*     wrb    = (bf16*)alloc((size_t)DD * DD * 2);
  bf16*     web    = (bf16*)alloc((size_t)DD * DD * 2);
  bf16*     wfcb   = (bf16*)alloc((size_t)DFF * DD * 2);
  bf16*     wprb   = (bf16*)alloc((size_t)DD * DFF * 2);
  bf16*     tbuf   = (bf16*)Rbase;  // N*512*2 = 51.2MB <= 54.4MB union
  bf16*     h2     = h_bf;
  float*    x2     = xr;

  (void)hipMemsetAsync(mb, 0, (size_t)NN * HH * 4, stream);
  (void)hipMemsetAsync(den, 0, (size_t)NN * HH * 4, stream);
  (void)hipMemsetAsync(attn, 0, (size_t)NN * DD * 4, stream);

  k_cvt_bf16<<<16, 256, 0, stream>>>(lin_l_w, wlb, DD * DD);
  k_cvt_bf16<<<16, 256, 0, stream>>>(lin_r_w, wrb, DD * DD);
  k_cvt_bf16<<<16, 256, 0, stream>>>(lin_e_w, web, DD * DD);
  k_cvt_bf16<<<64, 256, 0, stream>>>(fc_w, wfcb, DFF * DD);
  k_cvt_bf16<<<64, 256, 0, stream>>>(proj_w, wprb, DD * DFF);

  k_ln1<<<NN / 8, 256, 0, stream>>>(x, ln1_w, h_bf);
  k_xlxr<<<dim3(NN / 16, 2), 256, 0, stream>>>(h_bf, wlb, lin_l_b, xl, wrb, lin_r_b, xr);
  k_edge<<<EE / (16 * NT_EDGE), 256, 0, stream>>>(eattr, ei, ei + EE, web, xl, xr, att,
                                                  logits, mb);
  k_expden<<<EE * HH / 256, 256, 0, stream>>>(ei + EE, logits, mb, den);
  k_msg<<<EE * 32 / 256, 256, 0, stream>>>(ei, ei + EE, logits, den, xl, attn);
  k_resid_ln2<<<NN / 8, 256, 0, stream>>>(x, attn, attn_b, ln2_w, x2, h2);
  k_mlp1<<<NN / 16, 256, 0, stream>>>(h2, wfcb, tbuf);
  k_mlp2<<<NN / 16, 256, 0, stream>>>(tbuf, wprb, x2, out);
}
